// GRU_AcousticModel_52261162057796
// MI455X (gfx1250) — matrix operand order repaired and checked by reading, not compiled, not run on hardware
//
#include <hip/hip_runtime.h>
#include <hip/hip_bf16.h>
#include <math.h>

// ---------------------------------------------------------------------------
// GRU acoustic model on gfx1250 (wave32 + WMMA f32<=f16 16x16x32).
//   enc: 2048 sequential steps of [64x900]xK300 GEMM -> persistent kernel,
//        19 WGs (one per 16-wide hidden-column slice), 1 grid barrier/step.
//   fc : tiny [64x300] dense, VALU.
//   dec: hidden=13, single WG, LDS-resident state, WMMA per step.
// ---------------------------------------------------------------------------

#define FEAT   13
#define EMB    300
#define BATCH  64
#define SEQ    2048
#define KP     320          // hidden K padded to 10 * 32
#define NKS    10
#define ENC_WGS 19          // ceil(300 / 16)

typedef __attribute__((ext_vector_type(16))) _Float16 v16h;
typedef __attribute__((ext_vector_type(8)))  _Float16 h8;
typedef __attribute__((ext_vector_type(8)))  float    v8f;

union Frag { v16h v; h8 h[2]; };

__device__ __forceinline__ v8f wmma_f16(v16h a, v16h b, v8f c) {
  // D = A(16x32 f16) * B(32x16 f16) + C(16x16 f32)
  return __builtin_amdgcn_wmma_f32_16x16x32_f16(false, a, false, b,
                                                (short)0, c, false, false);
}

__device__ __forceinline__ float sigmoidf_(float x) {
  return 1.0f / (1.0f + __expf(-x));
}

// ---------------------------------------------------------------------------
// init: zero both hidden ping-pong buffers (incl. K padding) + barrier counter
// ---------------------------------------------------------------------------
__global__ void init_kernel(_Float16* __restrict__ hbuf, unsigned* __restrict__ bar) {
  int i = blockIdx.x * blockDim.x + threadIdx.x;
  if (i == 0) *bar = 0u;
  if (i < 2 * BATCH * KP) hbuf[i] = (_Float16)0.0f;
}

// ---------------------------------------------------------------------------
// encoder: persistent, 19 WGs x 128 threads (4 waves = 4 M-tiles of batch 64).
// WG b owns hidden columns [16b, 16b+16); computes r/z/n gate slices with
// WMMA, does the GRU cell update in registers, double-buffers h in global f16.
// ---------------------------------------------------------------------------
__global__ void __launch_bounds__(128, 1)
enc_kernel(const float* __restrict__ x,   const float* __restrict__ Wih,
           const float* __restrict__ Whh, const float* __restrict__ bih,
           const float* __restrict__ bhh, _Float16* __restrict__ hbuf,
           float* __restrict__ hfp, unsigned* __restrict__ bar)
{
  __shared__ _Float16 WhL[3][16][KP];   // hidden weights, f16, zero padded
  __shared__ _Float16 WiL[3][16][32];   // input  weights, f16, K 13->32

  const int tid  = threadIdx.x;
  const int lane = tid & 31;
  const int wave = tid >> 5;            // M-tile index (batch rows wave*16..+15)
  const int j0   = blockIdx.x * 16;     // hidden-column slice base

  // ---- stage this WG's weight slice into LDS (fp32 -> f16, zero padded) ----
  for (int idx = tid; idx < 3 * 16 * KP; idx += 128) {
    int g = idx / (16 * KP);
    int r = (idx / KP) & 15;
    int k = idx % KP;
    int row = j0 + r;
    float v = (row < EMB && k < EMB) ? Whh[(g * EMB + row) * EMB + k] : 0.0f;
    WhL[g][r][k] = (_Float16)v;
  }
  for (int idx = tid; idx < 3 * 16 * 32; idx += 128) {
    int g = idx / 512;
    int r = (idx >> 5) & 15;
    int k = idx & 31;
    int row = j0 + r;
    float v = (row < EMB && k < FEAT) ? Wih[(g * EMB + row) * FEAT + k] : 0.0f;
    WiL[g][r][k] = (_Float16)v;
  }
  __syncthreads();

  const int  n      = lane & 15;
  const int  hi     = (lane >> 4) & 1;
  const int  baseKA = hi * 8;           // 16-bit A-frag K split (0 / 8)
  const int  baseKB = hi * 8;           // 16-bit B-frag K split (0 / 8)
  const int  c      = j0 + n;           // hidden column this lane owns (C layout)
  const bool cvalid = (c < EMB);

  float br = 0.f, bz = 0.f, bi_n = 0.f, bh_n = 0.f;
  if (cvalid) {
    br   = bih[c]           + bhh[c];
    bz   = bih[EMB + c]     + bhh[EMB + c];
    bi_n = bih[2 * EMB + c];
    bh_n = bhh[2 * EMB + c];
  }

  // input-weight B fragments are time-invariant: keep in registers
  v16h bWi[3];
  #pragma unroll
  for (int g = 0; g < 3; ++g) {
    Frag f;
    f.h[0] = *(const h8*)&WiL[g][n][baseKB];
    f.h[1] = *(const h8*)&WiL[g][n][baseKB + 16];
    bWi[g] = f.v;
  }

  const int arow  = wave * 16 + n;       // batch row for A fragments
  const int mbase = wave * 16 + hi * 8;  // batch row base this lane owns in C
  const float* xr0 = x + (size_t)arow * SEQ * FEAT;

  float hprev[8];
  #pragma unroll
  for (int g = 0; g < 8; ++g) hprev[g] = 0.f;

  for (int t = 0; t < SEQ; ++t) {
    const _Float16* hrd = hbuf + (t & 1) * (BATCH * KP);        // h_{t-1}
    _Float16*       hwr = hbuf + ((t + 1) & 1) * (BATCH * KP);  // h_t

    // ---- input gates: A = x_t tile (16x32 f16, K 13 zero-padded) ----
    const float* xr = xr0 + t * FEAT;
    Frag fx;
    #pragma unroll
    for (int v = 0; v < 8; ++v) {
      int k = baseKA + ((v < 4) ? (2 * v) : (16 + 2 * (v - 4)));
      float f0 = (k     < FEAT) ? xr[k]     : 0.0f;
      float f1 = (k + 1 < FEAT) ? xr[k + 1] : 0.0f;
      fx.v[2 * v]     = (_Float16)f0;
      fx.v[2 * v + 1] = (_Float16)f1;
    }

    v8f accR = {}, accZ = {}, accIN = {}, accHN = {};
    accR  = wmma_f16(fx.v, bWi[0], accR);
    accZ  = wmma_f16(fx.v, bWi[1], accZ);
    accIN = wmma_f16(fx.v, bWi[2], accIN);

    // ---- hidden gates: gh = h_{t-1} @ Whh^T, K = 300 -> 10 x 32 ----
    #pragma unroll
    for (int ks = 0; ks < NKS; ++ks) {
      Frag fh;
      const _Float16* hp = hrd + arow * KP + ks * 32 + baseKA;
      fh.h[0] = *(const h8*)hp;
      fh.h[1] = *(const h8*)(hp + 16);
      Frag b0, b1, b2;
      const _Float16* w0 = &WhL[0][n][ks * 32 + baseKB];
      const _Float16* w1 = &WhL[1][n][ks * 32 + baseKB];
      const _Float16* w2 = &WhL[2][n][ks * 32 + baseKB];
      b0.h[0] = *(const h8*)w0; b0.h[1] = *(const h8*)(w0 + 16);
      b1.h[0] = *(const h8*)w1; b1.h[1] = *(const h8*)(w1 + 16);
      b2.h[0] = *(const h8*)w2; b2.h[1] = *(const h8*)(w2 + 16);
      accR  = wmma_f16(fh.v, b0.v, accR);
      accZ  = wmma_f16(fh.v, b1.v, accZ);
      accHN = wmma_f16(fh.v, b2.v, accHN);
    }

    // ---- GRU cell: lane owns rows mbase..mbase+7 at column c ----
    #pragma unroll
    for (int g = 0; g < 8; ++g) {
      float r  = sigmoidf_(accR[g] + br);
      float z  = sigmoidf_(accZ[g] + bz);
      float nn = tanhf(accIN[g] + bi_n + r * (accHN[g] + bh_n));
      hprev[g] = (1.0f - z) * nn + z * hprev[g];
    }

    if (cvalid) {
      #pragma unroll
      for (int g = 0; g < 8; ++g)
        hwr[(mbase + g) * KP + c] = (_Float16)hprev[g];
      if (t == SEQ - 1) {
        #pragma unroll
        for (int g = 0; g < 8; ++g)
          hfp[(mbase + g) * EMB + c] = hprev[g];
      }
    }

    // ---- one device-wide barrier per time step ----
    __threadfence();
    __syncthreads();
    if (tid == 0) {
      atomicAdd(bar, 1u);
      const unsigned tgt = (unsigned)(t + 1) * (unsigned)gridDim.x;
      volatile unsigned* vb = bar;
      while (*vb < tgt) __builtin_amdgcn_s_sleep(1);
    }
    __syncthreads();
    __threadfence();
  }
}

// ---------------------------------------------------------------------------
// fc: emb = relu(h_last @ fc_W^T + fc_b); also writes the emb output tail.
// ---------------------------------------------------------------------------
__global__ void fc_kernel(const float* __restrict__ hfp, const float* __restrict__ fcW,
                          const float* __restrict__ fcb, float* __restrict__ emb,
                          float* __restrict__ out_emb)
{
  int i = blockIdx.x * blockDim.x + threadIdx.x;
  if (i >= BATCH * EMB) return;
  int b = i / EMB, j = i % EMB;
  const float* hr = hfp + b * EMB;
  const float* wr = fcW + j * EMB;
  float s = fcb[j];
  for (int k = 0; k < EMB; ++k) s += hr[k] * wr[k];
  s = fmaxf(s, 0.0f);
  emb[i] = s;
  out_emb[i] = s;
}

// ---------------------------------------------------------------------------
// decoder: hidden = 13, constant input `emb`. Single WG, h state in LDS,
// dec_Whh B-fragments resident in registers, 3 WMMAs + cell update per step.
// ---------------------------------------------------------------------------
__global__ void __launch_bounds__(128, 1)
dec_kernel(const float* __restrict__ emb, const float* __restrict__ Wih,
           const float* __restrict__ Whh, const float* __restrict__ bih,
           const float* __restrict__ bhh, float* __restrict__ out)
{
  __shared__ _Float16 hd[BATCH][32];      // h state, K padded 13 -> 32
  __shared__ _Float16 WhL[3][16][32];

  const int tid  = threadIdx.x;
  const int lane = tid & 31;
  const int wave = tid >> 5;

  for (int idx = tid; idx < BATCH * 32; idx += 128)
    ((_Float16*)hd)[idx] = (_Float16)0.0f;
  for (int idx = tid; idx < 3 * 16 * 32; idx += 128) {
    int g = idx / 512;
    int r = (idx >> 5) & 15;
    int k = idx & 31;
    float v = (r < FEAT && k < FEAT) ? Whh[(g * FEAT + r) * FEAT + k] : 0.0f;
    WhL[g][r][k] = (_Float16)v;
  }
  __syncthreads();

  const int  n      = lane & 15;
  const int  hi     = (lane >> 4) & 1;
  const int  baseKA = hi * 8;
  const int  baseKB = hi * 8;
  const int  c      = n;
  const bool cvalid = (c < FEAT);
  const int  mbase  = wave * 16 + hi * 8;
  const int  arow   = wave * 16 + n;

  v16h bW[3];
  #pragma unroll
  for (int g = 0; g < 3; ++g) {
    Frag f;
    f.h[0] = *(const h8*)&WhL[g][n][baseKB];
    f.h[1] = *(const h8*)&WhL[g][n][baseKB + 16];
    bW[g] = f.v;
  }

  // constant input gates dgi = emb @ dec_Wih^T + dec_bih (one-time, VALU)
  float bhr = 0.f, bhz = 0.f, bhn = 0.f;
  float giR[8], giZ[8], giN[8];
  #pragma unroll
  for (int g = 0; g < 8; ++g) { giR[g] = giZ[g] = giN[g] = 0.f; }
  if (cvalid) {
    bhr = bhh[c]; bhz = bhh[FEAT + c]; bhn = bhh[2 * FEAT + c];
    #pragma unroll
    for (int g = 0; g < 8; ++g) {
      giR[g] = bih[c]; giZ[g] = bih[FEAT + c]; giN[g] = bih[2 * FEAT + c];
    }
    for (int k = 0; k < EMB; ++k) {
      float wr = Wih[c * EMB + k];
      float wz = Wih[(FEAT + c) * EMB + k];
      float wn = Wih[(2 * FEAT + c) * EMB + k];
      #pragma unroll
      for (int g = 0; g < 8; ++g) {
        float e = emb[(mbase + g) * EMB + k];
        giR[g] += e * wr; giZ[g] += e * wz; giN[g] += e * wn;
      }
    }
  }

  float hprev[8];
  #pragma unroll
  for (int g = 0; g < 8; ++g) hprev[g] = 0.f;

  for (int t = 0; t < SEQ; ++t) {
    Frag fh;
    fh.h[0] = *(const h8*)&hd[arow][baseKA];
    fh.h[1] = *(const h8*)&hd[arow][baseKA + 16];
    v8f aR = {}, aZ = {}, aH = {};
    aR = wmma_f16(fh.v, bW[0], aR);
    aZ = wmma_f16(fh.v, bW[1], aZ);
    aH = wmma_f16(fh.v, bW[2], aH);
    __syncthreads();                       // all reads of hd done
    #pragma unroll
    for (int g = 0; g < 8; ++g) {
      float r  = sigmoidf_(giR[g] + aR[g] + bhr);
      float z  = sigmoidf_(giZ[g] + aZ[g] + bhz);
      float nn = tanhf(giN[g] + r * (aH[g] + bhn));
      hprev[g] = (1.0f - z) * nn + z * hprev[g];
    }
    if (cvalid) {
      #pragma unroll
      for (int g = 0; g < 8; ++g) {
        int b = mbase + g;
        out[(size_t)b * SEQ * FEAT + (size_t)t * FEAT + c] = hprev[g];
        hd[b][c] = (_Float16)hprev[g];
      }
    }
    __syncthreads();                       // h_t visible for next step
  }
}

// ---------------------------------------------------------------------------
extern "C" void kernel_launch(void* const* d_in, const int* in_sizes, int n_in,
                              void* d_out, int out_size, void* d_ws, size_t ws_size,
                              hipStream_t stream) {
  (void)in_sizes; (void)n_in; (void)out_size; (void)ws_size;
  const float* x    = (const float*)d_in[0];
  const float* eWih = (const float*)d_in[1];
  const float* eWhh = (const float*)d_in[2];
  const float* ebih = (const float*)d_in[3];
  const float* ebhh = (const float*)d_in[4];
  const float* fcW  = (const float*)d_in[5];
  const float* fcb  = (const float*)d_in[6];
  const float* dWih = (const float*)d_in[7];
  const float* dWhh = (const float*)d_in[8];
  const float* dbih = (const float*)d_in[9];
  const float* dbhh = (const float*)d_in[10];
  float* out = (float*)d_out;

  // workspace layout (all offsets 256B-aligned)
  char* w = (char*)d_ws;
  _Float16* hbuf = (_Float16*)(w);                     // 2 x 64 x 320 f16 = 81920 B
  float*    hfp  = (float*)(w + 81920);                // 64 x 300 f32     = 76800 B
  float*    embb = (float*)(w + 81920 + 76800);        // 64 x 300 f32     = 76800 B
  unsigned* bar  = (unsigned*)(w + 81920 + 76800 + 76800);

  init_kernel<<<(2 * BATCH * KP + 255) / 256, 256, 0, stream>>>(hbuf, bar);
  enc_kernel<<<ENC_WGS, 128, 0, stream>>>(x, eWih, eWhh, ebih, ebhh, hbuf, hfp, bar);
  fc_kernel<<<(BATCH * EMB + 255) / 256, 256, 0, stream>>>(
      hfp, fcW, fcb, embb, out + (size_t)BATCH * SEQ * FEAT);
  dec_kernel<<<1, 128, 0, stream>>>(embb, dWih, dWhh, dbih, dbhh, out);
}
